// AttentionResBlock_25881472925730
// MI455X (gfx1250) — matrix operand order repaired and checked by reading, not compiled, not run on hardware
//
#include <hip/hip_runtime.h>

// ---------------------------------------------------------------------------
// Problem constants (from reference): B=4, T=4096, C=256, window=512, nw=8.
// Strategy: bf16 WMMA (v_wmma_f32_16x16x32_bf16) for QK^T, P@V and both
// 1x1-conv GEMMs; f32 softmax + gate on VALU. All WMMA fragment loads are
// contiguous 16B chunks per lane (k = 16*(lane/16) + s, same order in A and B).
// ---------------------------------------------------------------------------

#define BB   4
#define TT   4096
#define CC   256
#define WW   512
#define NW   8

typedef __attribute__((ext_vector_type(16))) __bf16        v16bf;
typedef __attribute__((ext_vector_type(8)))  float         v8f;
typedef __attribute__((ext_vector_type(4)))  unsigned int  u32x4;

union ABFrag {            // 16 bf16 = 32 bytes = two 16B vector loads
    v16bf v;
    u32x4 q[2];
};

__device__ __forceinline__ unsigned short f2bf(float f) {
    unsigned u = __float_as_uint(f);
    unsigned r = u + 0x7FFFu + ((u >> 16) & 1u);   // round-to-nearest-even
    return (unsigned short)(r >> 16);
}

// LDS geometry for the attention kernel (all row strides keep 16B alignment)
#define SSTR 1032                       // f32 score row stride (16 x 1024)
#define PSTR 1040                       // bf16 prob row stride
#define VSTR 136                        // bf16 Vt row stride (256ch x 128keys)
#define SC_BYTES (16 * SSTR * 4)        // 66048
#define PB_BYTES (16 * PSTR * 2)        // 33280
#define VT_BYTES (256 * VSTR * 2)       // 69632
#define SMEM_TOTAL (SC_BYTES + PB_BYTES + VT_BYTES)   // 168960 < 320KB/WGP

// ---------------------------------------------------------------------------
// Kernel 1: f32 -> bf16 conversion of x, Wr, Ws into workspace.
// ---------------------------------------------------------------------------
__global__ void cvt_kernel(const float* __restrict__ x,
                           const float* __restrict__ Wr,
                           const float* __restrict__ Ws,
                           unsigned short* __restrict__ xb,
                           unsigned short* __restrict__ wrb,
                           unsigned short* __restrict__ wsb,
                           int nx, int nw) {
    int total = nx + 2 * nw;
    for (int i = blockIdx.x * blockDim.x + threadIdx.x; i < total;
         i += gridDim.x * blockDim.x) {
        if (i < nx)               xb[i]           = f2bf(x[i]);
        else if (i < nx + nw)     wrb[i - nx]     = f2bf(Wr[i - nx]);
        else                      wsb[i - nx - nw] = f2bf(Ws[i - nx - nw]);
    }
}

// ---------------------------------------------------------------------------
// Kernel 2: local causal attention + sigmoid*tanh gate. One block handles
// 16 queries of one (batch, window): 8 waves.
//   Phase 1: scores S (16x1024) via WMMA, masked, f32 -> LDS.
//   Phase 2: softmax over 1024 keys (16 lanes/row, shuffle reductions).
//   Phase 3: O = P @ V via WMMA over 128-key V chunks transposed in LDS.
// ---------------------------------------------------------------------------
__global__ __launch_bounds__(256) void attn_kernel(
        const unsigned short* __restrict__ xb,
        unsigned short* __restrict__ ob) {
    extern __shared__ char smem[];
    float*          sc = (float*)smem;
    unsigned short* pb = (unsigned short*)(smem + SC_BYTES);
    unsigned short* vt = (unsigned short*)(smem + SC_BYTES + PB_BYTES);

    const int tid  = threadIdx.x;
    const int wave = tid >> 5;
    const int lane = tid & 31;
    const int lrow = lane & 15;
    const int sel  = lane >> 4;          // 0: lanes 0-15, 1: lanes 16-31

    const int bid = blockIdx.x;
    const int b   = bid / (NW * 32);
    const int rem = bid % (NW * 32);
    const int win = rem / 32;
    const int qt  = rem % 32;            // 16-query tile within window

    const long qrow = (long)b * TT + (long)win * WW + qt * 16;
    const long kv0  = (long)b * TT + (long)(win - 1) * WW;  // key j=0 row
    const long bmin = (long)b * TT;

    // ---------------- Phase 1: scores --------------------------------------
    const int nbase = wave * 128;        // this wave's 128-key slab
    const unsigned short* qptr = xb + (qrow + lrow) * CC;
    for (int nt = 0; nt < 8; ++nt) {
        const int j = nbase + nt * 16 + lrow;       // key column (0..1023)
        long krow = kv0 + j;
        if (krow < bmin) krow = bmin;               // masked later anyway
        const unsigned short* kptr = xb + krow * CC;

        v8f acc = {0.f, 0.f, 0.f, 0.f, 0.f, 0.f, 0.f, 0.f};
#pragma unroll
        for (int ks = 0; ks < 8; ++ks) {
            ABFrag a, bm;
            // A (Q 16x32): lane<16 -> k 0..15; lane>=16 -> k 16..31
            const unsigned short* pa = qptr + ks * 32 + sel * 16;
            a.q[0] = *(const u32x4*)(pa);
            a.q[1] = *(const u32x4*)(pa + 8);
            // B (K^T 32x16): lane<16 -> k 0..15; lane>=16 -> k 16..31
            const unsigned short* pk = kptr + ks * 32 + sel * 16;
            bm.q[0] = *(const u32x4*)(pk);
            bm.q[1] = *(const u32x4*)(pk + 8);
            acc = __builtin_amdgcn_wmma_f32_16x16x32_bf16(
                      false, a.v, false, bm.v, (short)0, acc, false, false);
        }
#pragma unroll
        for (int r = 0; r < 8; ++r) {
            const int M  = r + sel * 8;             // query row in tile
            const int iq = qt * 16 + M;             // query idx in window
            const bool valid = (j <= iq + WW) && ((j >= WW) || (win > 0));
            sc[M * SSTR + j] = valid ? acc[r] * 0.0625f : -1e9f;  // *C^-0.5
        }
    }
    __syncthreads();

    // ---------------- Phase 2: softmax -------------------------------------
    {
        const int rowi = tid >> 4;                  // 0..15
        const int ch   = tid & 15;                  // 64-col chunk
        float* rp = sc + rowi * SSTR + ch * 64;
        float m = -1e30f;
#pragma unroll 4
        for (int k = 0; k < 64; ++k) m = fmaxf(m, rp[k]);
        for (int off = 8; off; off >>= 1) m = fmaxf(m, __shfl_xor(m, off, 16));
        float s = 0.f;
#pragma unroll 4
        for (int k = 0; k < 64; ++k) s += __expf(rp[k] - m);
        for (int off = 8; off; off >>= 1) s += __shfl_xor(s, off, 16);
        const float inv = 1.0f / s;
        unsigned short* pp = pb + rowi * PSTR + ch * 64;
#pragma unroll 4
        for (int k = 0; k < 64; ++k) pp[k] = f2bf(__expf(rp[k] - m) * inv);
    }

    // ---------------- Phase 3: O = P @ V -----------------------------------
    const int ct0 = wave * 2;                       // 2 channel tiles / wave
    v8f o0 = {0.f, 0.f, 0.f, 0.f, 0.f, 0.f, 0.f, 0.f};
    v8f o1 = {0.f, 0.f, 0.f, 0.f, 0.f, 0.f, 0.f, 0.f};

    for (int c = 0; c < 8; ++c) {                   // 128-key chunks
        __syncthreads();                            // protect vt reuse / pb
        {   // stage V^T chunk: vt[ch][keyLocal]
            const int keyl = tid >> 1;
            const int chb  = (tid & 1) * 128;
            long krow = kv0 + c * 128 + keyl;
            if (krow < bmin) krow = bmin;           // P==0 there
            const unsigned short* src = xb + krow * CC + chb;
#pragma unroll
            for (int c8 = 0; c8 < 128; c8 += 8) {
                union { u32x4 q; unsigned short s[8]; } t;
                t.q = *(const u32x4*)(src + c8);
#pragma unroll
                for (int e = 0; e < 8; ++e)
                    vt[(chb + c8 + e) * VSTR + keyl] = t.s[e];
            }
        }
        __syncthreads();
#pragma unroll
        for (int ks = 0; ks < 4; ++ks) {
            ABFrag pA, vB0, vB1;
            const unsigned short* pa =
                pb + lrow * PSTR + c * 128 + ks * 32 + sel * 16;
            pA.q[0] = *(const u32x4*)(pa);
            pA.q[1] = *(const u32x4*)(pa + 8);
            const unsigned short* pv0 =
                vt + (ct0 * 16 + lrow) * VSTR + ks * 32 + sel * 16;
            vB0.q[0] = *(const u32x4*)(pv0);
            vB0.q[1] = *(const u32x4*)(pv0 + 8);
            const unsigned short* pv1 = pv0 + 16 * VSTR;
            vB1.q[0] = *(const u32x4*)(pv1);
            vB1.q[1] = *(const u32x4*)(pv1 + 8);
            o0 = __builtin_amdgcn_wmma_f32_16x16x32_bf16(
                     false, pA.v, false, vB0.v, (short)0, o0, false, false);
            o1 = __builtin_amdgcn_wmma_f32_16x16x32_bf16(
                     false, pA.v, false, vB1.v, (short)0, o1, false, false);
        }
    }

    // gate: sigmoid(a)*tanh(a), store bf16 o
#pragma unroll
    for (int r = 0; r < 8; ++r) {
        const int  M    = r + sel * 8;
        const long orow = qrow + M;
        float a0 = o0[r];
        float a1 = o1[r];
        float g0 = (1.f / (1.f + __expf(-a0))) * tanhf(a0);
        float g1 = (1.f / (1.f + __expf(-a1))) * tanhf(a1);
        ob[orow * CC + ct0 * 16 + lrow]        = f2bf(g0);
        ob[orow * CC + (ct0 + 1) * 16 + lrow]  = f2bf(g1);
    }
}

// ---------------------------------------------------------------------------
// Kernel 3: both projections as one GEMM, N=512 (residual | skip).
// Block = 64 rows x 512 cols, 8 waves; each wave: 4 M-tiles x 4 N-tiles,
// A fragments cached in registers per M-tile (8 x v16bf).
// ---------------------------------------------------------------------------
__global__ __launch_bounds__(256) void proj_kernel(
        const unsigned short* __restrict__ ob,
        const unsigned short* __restrict__ wrb,
        const unsigned short* __restrict__ wsb,
        const float* __restrict__ br,
        const float* __restrict__ bs,
        float* __restrict__ outR,
        float* __restrict__ outS) {
    const int tid  = threadIdx.x;
    const int wave = tid >> 5;
    const int lane = tid & 31;
    const int lrow = lane & 15;
    const int sel  = lane >> 4;
    const long rb  = (long)blockIdx.x * 64;

    for (int mt = 0; mt < 4; ++mt) {
        ABFrag a[8];
        const unsigned short* arow = ob + (rb + mt * 16 + lrow) * CC;
#pragma unroll
        for (int ks = 0; ks < 8; ++ks) {
            const unsigned short* pa = arow + ks * 32 + sel * 16;
            a[ks].q[0] = *(const u32x4*)(pa);
            a[ks].q[1] = *(const u32x4*)(pa + 8);
        }
#pragma unroll
        for (int nl = 0; nl < 4; ++nl) {
            const int  ct  = wave * 4 + nl;          // 0..31 (concat N)
            const bool isR = ct < 16;
            const int  dt  = isR ? ct : ct - 16;
            const int  d   = dt * 16 + lrow;         // output channel (B col)
            const unsigned short* wrow = (isR ? wrb : wsb) + d * CC;
            v8f acc = {0.f, 0.f, 0.f, 0.f, 0.f, 0.f, 0.f, 0.f};
#pragma unroll
            for (int ks = 0; ks < 8; ++ks) {
                ABFrag bm;
                const unsigned short* pw = wrow + ks * 32 + sel * 16;
                bm.q[0] = *(const u32x4*)(pw);
                bm.q[1] = *(const u32x4*)(pw + 8);
                acc = __builtin_amdgcn_wmma_f32_16x16x32_bf16(
                          false, a[ks].v, false, bm.v, (short)0, acc,
                          false, false);
            }
            const float bias = isR ? br[d] : bs[d];
            float* outp = isR ? outR : outS;
#pragma unroll
            for (int r = 0; r < 8; ++r) {
                const long row = rb + mt * 16 + r + sel * 8;
                outp[row * CC + d] = acc[r] + bias;
            }
        }
    }
}

// ---------------------------------------------------------------------------
extern "C" void kernel_launch(void* const* d_in, const int* in_sizes, int n_in,
                              void* d_out, int out_size, void* d_ws,
                              size_t ws_size, hipStream_t stream) {
    const float* x  = (const float*)d_in[0];
    const float* Wr = (const float*)d_in[1];
    const float* br = (const float*)d_in[2];
    const float* Ws = (const float*)d_in[3];
    const float* bs = (const float*)d_in[4];

    const int NX = BB * TT * CC;          // 4,194,304
    const int NWGT = CC * CC;             // 65,536

    unsigned short* xb  = (unsigned short*)d_ws;          // 8 MB
    unsigned short* obuf = xb + NX;                       // 8 MB
    unsigned short* wrb = obuf + NX;                      // 128 KB
    unsigned short* wsb = wrb + NWGT;                     // 128 KB

    float* outR = (float*)d_out;
    float* outS = outR + (long)BB * TT * CC;

    cvt_kernel<<<4096, 256, 0, stream>>>(x, Wr, Ws, xb, wrb, wsb, NX, NWGT);
    attn_kernel<<<BB * NW * 32, 256, SMEM_TOTAL, stream>>>(xb, obuf);
    proj_kernel<<<(BB * TT) / 64, 256, 0, stream>>>(obuf, wrb, wsb, br, bs,
                                                    outR, outS);
}
